// CogKR_62594853372415
// MI455X (gfx1250) — matrix operand order repaired and checked by reading, not compiled, not run on hardware
//
#include <hip/hip_runtime.h>
#include <hip/hip_bf16.h>
#include <stdint.h>

typedef __attribute__((ext_vector_type(16))) _Float16 v16h;
typedef __attribute__((ext_vector_type(8)))  float    v8f;

#define B_    256
#define K_    32
#define M_    64
#define MN_   256
#define E_    128
#define H_    128
#define Q_    128

// f16 weight table offsets (halves) inside ws f16 region
#define W16_HID   0        // hidden_W   (128x128) 16384
#define W16_PASS  16384    // pass_W     (128x256) 32768
#define W16_NEXT  49152    // nexthop_W  (128x256) 32768
#define W16_CAND  81920    // candidate_W(128x384) 49152
#define W16_TOTAL 131072

__device__ __forceinline__ float lrelu(float x) { return x > 0.0f ? x : 0.01f * x; }

__device__ __forceinline__ v8f wmma32f16(v16h a, v16h b, v8f c) {
  return __builtin_amdgcn_wmma_f32_16x16x32_f16(false, a, false, b, (short)0, c,
                                                false, false);
}

// B-fragment from a pre-converted f16 weight table, row-major (n, k), stride ldw.
// lane L: n = n0 + (L&15), K half = (L>=16)*16 -> 16 contiguous halves (32B).
__device__ __forceinline__ v16h load_bfrag16(const _Float16* __restrict__ W, int ldw,
                                             int n0, int k0) {
  int lane = threadIdx.x & 31;
  const _Float16* p = W + (size_t)(n0 + (lane & 15)) * ldw + k0 + ((lane >> 4) << 4);
  return *(const v16h*)p;
}

// ------------------------------------------------------------------
// K0: convert the four weight matrices f32 -> f16 (once, 131072 elems)
// ------------------------------------------------------------------
__global__ void k_cvt_w(const float* __restrict__ hW, const float* __restrict__ pW,
                        const float* __restrict__ nW, const float* __restrict__ cW,
                        _Float16* __restrict__ out) {
  int i = blockIdx.x * 256 + threadIdx.x;
  float v;
  if (i < 16384)       v = hW[i];
  else if (i < 49152)  v = pW[i - 16384];
  else if (i < 81920)  v = nW[i - 49152];
  else                 v = cW[i - 81920];
  out[i] = (_Float16)v;
}

// ------------------------------------------------------------------
// K1: init = lrelu(entity_emb[start] @ hidden_W.T + hidden_b)  (256x128)
// grid (B/16, H/16), block 32
// ------------------------------------------------------------------
__global__ void k_init(const int* __restrict__ start, const float* __restrict__ eemb,
                       const _Float16* __restrict__ hW16, const float* __restrict__ hb,
                       float* __restrict__ init_out) {
  int rowBase = blockIdx.x * 16, colBase = blockIdx.y * 16;
  int lane = threadIdx.x & 31;
  int mm = lane & 15;
  int ko = (lane >> 4) * 16;
  const float* arow = eemb + (size_t)start[rowBase + mm] * E_;
  v8f c = {};
  for (int k0 = 0; k0 < E_; k0 += 32) {
    v16h a;
#pragma unroll
    for (int i = 0; i < 8; ++i) {
      a[i]     = (_Float16)arow[k0 + ko + i];
      a[8 + i] = (_Float16)arow[k0 + ko + 8 + i];
    }
    v16h b = load_bfrag16(hW16, E_, colBase, k0);
    c = wmma32f16(a, b, c);
  }
  int n = colBase + mm;
  float bias = hb[n];
#pragma unroll
  for (int r = 0; r < 8; ++r) {
    int R = rowBase + r + ((lane >> 4) << 3);
    init_out[(size_t)R * H_ + n] = lrelu(c[r] + bias);
  }
}

// ------------------------------------------------------------------
// K2a: per (b,k): relsum/denom, cnt0/denom, cnt/denom
// grid (B*K), block 128
// ------------------------------------------------------------------
__global__ void k_agg_gather(const int* __restrict__ nb_nodes,
                             const int* __restrict__ nb_rels,
                             const int* __restrict__ nb_num,
                             const float* __restrict__ remb,
                             float* __restrict__ relsum, float* __restrict__ s0n,
                             float* __restrict__ cntd) {
  __shared__ int s_rel[M_];
  __shared__ int s_zero[M_];
  int row = blockIdx.x;  // b*K + k
  int t = threadIdx.x;   // 0..127 (channel)
  int cnt = nb_num[row];
  const int* nn = nb_nodes + (size_t)row * M_;
  const int* nr = nb_rels + (size_t)row * M_;
  if (t < M_) {
    s_rel[t] = nr[t];
    s_zero[t] = (t < cnt && nn[t] == 0) ? 1 : 0;
  }
  __syncthreads();
  float inv = 1.0f / (float)(cnt ? cnt : 1);
  float acc = 0.0f;
  for (int m = 0; m < M_; ++m)
    if (m < cnt) acc += remb[(size_t)s_rel[m] * E_ + t];
  relsum[(size_t)row * E_ + t] = acc * inv;
  if (t == 0) {
    int c0 = 0;
    for (int m = 0; m < M_; ++m) c0 += s_zero[m];
    s0n[row] = (float)c0 * inv;
    cntd[row] = (float)cnt * inv;
  }
}

// ------------------------------------------------------------------
// K2b: upd = lrelu([aim_e | init*s0n | relsum] @ [hidden_W | pass_W].T
//                  + hidden_b + cntd*pass_b)       (8192 x 384 x 128)
// grid (B*K/16, H/16), block 32
// ------------------------------------------------------------------
__global__ void k_agg_gemm(const int* __restrict__ aims, const float* __restrict__ eemb,
                           const float* __restrict__ init, const float* __restrict__ relsum,
                           const float* __restrict__ s0n, const float* __restrict__ cntd,
                           const _Float16* __restrict__ hW16,
                           const _Float16* __restrict__ pW16,
                           const float* __restrict__ hb, const float* __restrict__ pb,
                           float* __restrict__ upd) {
  int rowBase = blockIdx.x * 16, colBase = blockIdx.y * 16;
  int lane = threadIdx.x & 31;
  int mm = lane & 15;
  int ko = (lane >> 4) * 16;
  int row = rowBase + mm;
  int b = row >> 5;  // row / K_
  const float* aimrow = eemb + (size_t)aims[row] * E_;
  const float* initrow = init + (size_t)b * H_;
  const float* relrow = relsum + (size_t)row * E_;
  float s0 = s0n[row];
  v8f c = {};
  for (int kc = 0; kc < 12; ++kc) {
    int k0 = kc * 32;  // uniform; segment boundaries align to 128
    v16h a;
#pragma unroll
    for (int i = 0; i < 8; ++i) {
      int ka = k0 + ko + i, kb = k0 + ko + 8 + i;
      float va, vb;
      if (k0 < 128)      { va = aimrow[ka];             vb = aimrow[kb]; }
      else if (k0 < 256) { va = initrow[ka - 128] * s0; vb = initrow[kb - 128] * s0; }
      else               { va = relrow[ka - 256];       vb = relrow[kb - 256]; }
      a[i] = (_Float16)va; a[8 + i] = (_Float16)vb;
    }
    int kbb = k0 + ((lane >> 4) << 4);
    int n = colBase + mm;
    const _Float16* p = (k0 < 128) ? (hW16 + (size_t)n * E_ + kbb)
                                   : (pW16 + (size_t)n * 256 + (kbb - 128));
    v16h bb = *(const v16h*)p;
    c = wmma32f16(a, bb, c);
  }
  int n = colBase + mm;
  float bh = hb[n], bp = pb[n];
#pragma unroll
  for (int r = 0; r < 8; ++r) {
    int R = rowBase + r + ((lane >> 4) << 3);
    upd[(size_t)R * H_ + n] = lrelu(c[r] + bh + cntd[R] * bp);
  }
}

// ------------------------------------------------------------------
// K3a: cur_cat = [node_gather(currents) | query]   (256 x 256)
// ------------------------------------------------------------------
__global__ void k_curcat(const int* __restrict__ currents, const float* __restrict__ init,
                         const float* __restrict__ upd, const float* __restrict__ query,
                         float* __restrict__ curcat) {
  int b = blockIdx.x, t = threadIdx.x;
  float v;
  if (t < H_) {
    int idx = currents[b];
    if (idx == 0)       v = init[(size_t)b * H_ + t];
    else if (idx <= K_) v = upd[((size_t)b * K_ + (idx - 1)) * H_ + t];
    else                v = 0.0f;
  } else {
    v = query[(size_t)b * Q_ + (t - H_)];
  }
  curcat[(size_t)b * (H_ + Q_) + t] = v;
}

// ------------------------------------------------------------------
// K3b: state = lrelu(cur_cat @ nexthop_W.T + nexthop_b)  (256x256x128)
// grid (B/16, H/16), block 32
// ------------------------------------------------------------------
__global__ void k_state_gemm(const float* __restrict__ A,
                             const _Float16* __restrict__ nW16,
                             const float* __restrict__ bias, float* __restrict__ out) {
  int rowBase = blockIdx.x * 16, colBase = blockIdx.y * 16;
  int lane = threadIdx.x & 31;
  int mm = lane & 15;
  int ko = (lane >> 4) * 16;
  const float* arow = A + (size_t)(rowBase + mm) * 256;
  v8f c = {};
  for (int k0 = 0; k0 < 256; k0 += 32) {
    v16h a;
#pragma unroll
    for (int i = 0; i < 8; ++i) {
      a[i]     = (_Float16)arow[k0 + ko + i];
      a[8 + i] = (_Float16)arow[k0 + ko + 8 + i];
    }
    v16h b = load_bfrag16(nW16, 256, colBase, k0);
    c = wmma32f16(a, b, c);
  }
  int n = colBase + mm;
  float bv = bias[n];
#pragma unroll
  for (int r = 0; r < 8; ++r) {
    int R = rowBase + r + ((lane >> 4) << 3);
    out[(size_t)R * H_ + n] = lrelu(c[r] + bv);
  }
}

// ------------------------------------------------------------------
// K3c: thresh[b] = cur_cat[b] . gate_W + gate_b -> d_out[b*257 + 256]
// ------------------------------------------------------------------
__global__ void k_gate(const float* __restrict__ curcat, const float* __restrict__ gW,
                       const float* __restrict__ gb, float* __restrict__ outp) {
  int b = threadIdx.x;  // 256 threads
  const float* x = curcat + (size_t)b * 256;
  float acc = 0.0f;
  for (int j = 0; j < 256; ++j) acc += x[j] * gW[j];
  outp[(size_t)b * (MN_ + 1) + MN_] = acc + gb[0];
}

// ------------------------------------------------------------------
// K4: candidate GEMM + lrelu + dot(state) + mask -> scores.
// Block = 256 threads (8 waves), covers 128 candidates of one b.
// candidate_W (f16, 96KB) staged to LDS once per block via
// global_load_async_to_lds_b128 (ASYNCcnt), shared by all 8 waves.
// grid (MN/128, B), block 256
// ------------------------------------------------------------------
__global__ void k_cand(const int* __restrict__ cnodes, const int* __restrict__ cents,
                       const int* __restrict__ crels, const int* __restrict__ cmask,
                       const float* __restrict__ init, const float* __restrict__ upd,
                       const float* __restrict__ eemb, const float* __restrict__ remb,
                       const _Float16* __restrict__ cW16, const float* __restrict__ cb,
                       const float* __restrict__ state, float* __restrict__ outp) {
  __shared__ _Float16 sW[H_ * 384];  // 96 KB: candidate_W in f16, [n][k]
  int b = blockIdx.y;
  int t = threadIdx.x;
  int w = t >> 5;           // wave id 0..7
  int lane = t & 31;
  int mm = lane & 15;
  int ko = (lane >> 4) * 16;
  int m0 = blockIdx.x * 128 + w * 16;

  // ---- async stage candidate_W(f16) -> LDS (24 x b128 per thread) ----
  {
    uint32_t lbase = (uint32_t)(uintptr_t)&sW[0];
    uint64_t gbase = (uint64_t)(uintptr_t)cW16;
#pragma unroll
    for (int i = 0; i < 24; ++i) {
      uint32_t off = (uint32_t)((i * 256 + t) * 16);
      uint32_t la = lbase + off;
      uint64_t ga = gbase + off;
      asm volatile("global_load_async_to_lds_b128 %0, %1, off"
                   :: "v"(la), "v"(ga) : "memory");
    }
    asm volatile("s_wait_asynccnt 0x0" ::: "memory");
  }
  __syncthreads();

  int cidx = m0 + mm;
  int node = cnodes[(size_t)b * MN_ + cidx];
  float nscale = 1.0f;
  const float* nodeptr;
  if (node == 0)       nodeptr = init + (size_t)b * H_;
  else if (node <= K_) nodeptr = upd + ((size_t)b * K_ + (node - 1)) * H_;
  else                 { nodeptr = init + (size_t)b * H_; nscale = 0.0f; }
  const float* entrow = eemb + (size_t)cents[(size_t)b * MN_ + cidx] * E_;
  const float* relrow = remb + (size_t)crels[(size_t)b * MN_ + cidx] * E_;

  v8f acc[8] = {};
  for (int kc = 0; kc < 12; ++kc) {
    int k0 = kc * 32;
    v16h a;
#pragma unroll
    for (int i = 0; i < 8; ++i) {
      int ka = k0 + ko + i, kb = k0 + ko + 8 + i;
      float va, vb;
      if (k0 < 128)      { va = nodeptr[ka] * nscale; vb = nodeptr[kb] * nscale; }
      else if (k0 < 256) { va = entrow[ka - 128];     vb = entrow[kb - 128]; }
      else               { va = relrow[ka - 256];     vb = relrow[kb - 256]; }
      a[i] = (_Float16)va; a[8 + i] = (_Float16)vb;
    }
    int kbb = k0 + ((lane >> 4) << 4);
#pragma unroll
    for (int nt = 0; nt < 8; ++nt) {
      int n = nt * 16 + mm;
      v16h bb = *(const v16h*)&sW[(size_t)n * 384 + kbb];  // ds_load path
      acc[nt] = wmma32f16(a, bb, acc[nt]);
    }
  }

  // ---- epilogue: lrelu, dot with state over this lane's columns ----
  const float* st = state + (size_t)b * H_;
  float part[8];
#pragma unroll
  for (int r = 0; r < 8; ++r) part[r] = 0.0f;
#pragma unroll
  for (int nt = 0; nt < 8; ++nt) {
    int n = nt * 16 + mm;
    float bias = cb[n], sv = st[n];
#pragma unroll
    for (int r = 0; r < 8; ++r) part[r] += lrelu(acc[nt][r] + bias) * sv;
  }

  // reuse weight LDS as reduction scratch (all waves done reading sW)
  __syncthreads();
  float* s_red = (float*)sW;  // per wave: 32 lanes x 8 partials
#pragma unroll
  for (int r = 0; r < 8; ++r) s_red[(w * 32 + lane) * 8 + r] = part[r];
  __syncthreads();
  if (lane < 16) {
    int rowM = lane;
    int base = (rowM >= 8) ? 16 : 0;
    int r = rowM & 7;
    float s = 0.0f;
#pragma unroll
    for (int j = 0; j < 16; ++j) s += s_red[(w * 32 + base + j) * 8 + r];
    s *= 0.08838834764831845f;  // 1/sqrt(E), E=128
    if (!cmask[(size_t)b * MN_ + m0 + rowM]) s = -100000.0f;
    outp[(size_t)b * (MN_ + 1) + m0 + rowM] = s;
  }
}

// ------------------------------------------------------------------
extern "C" void kernel_launch(void* const* d_in, const int* in_sizes, int n_in,
                              void* d_out, int out_size, void* d_ws, size_t ws_size,
                              hipStream_t stream) {
  const int* start_entities      = (const int*)d_in[0];
  const int* aims                = (const int*)d_in[1];
  // d_in[2] = node_pos (deterministically arange(1,K+1); folded into indexing)
  const int* neighbor_nodes      = (const int*)d_in[3];
  const int* neighbor_relations  = (const int*)d_in[4];
  const int* neighbors_num       = (const int*)d_in[5];
  const int* currents            = (const int*)d_in[6];
  const int* candidate_nodes     = (const int*)d_in[7];
  const int* candidate_entities  = (const int*)d_in[8];
  const int* candidate_relations = (const int*)d_in[9];
  const int* candidate_masks     = (const int*)d_in[10];
  const float* entity_emb   = (const float*)d_in[11];
  const float* relation_emb = (const float*)d_in[12];
  const float* hidden_W     = (const float*)d_in[13];
  const float* hidden_b     = (const float*)d_in[14];
  const float* pass_W       = (const float*)d_in[15];
  const float* pass_b       = (const float*)d_in[16];
  const float* nexthop_W    = (const float*)d_in[17];
  const float* nexthop_b    = (const float*)d_in[18];
  const float* candidate_W  = (const float*)d_in[19];
  const float* candidate_b  = (const float*)d_in[20];
  const float* gate_W       = (const float*)d_in[21];
  const float* gate_b       = (const float*)d_in[22];
  const float* query        = (const float*)d_in[23];
  float* out = (float*)d_out;

  float* ws = (float*)d_ws;
  float* w_init   = ws;                               // B*H      = 32768
  float* w_upd    = w_init + (size_t)B_ * H_;         // B*K*H    = 1048576
  float* w_relsum = w_upd + (size_t)B_ * K_ * H_;     // B*K*E    = 1048576
  float* w_s0n    = w_relsum + (size_t)B_ * K_ * E_;  // B*K      = 8192
  float* w_cntd   = w_s0n + (size_t)B_ * K_;          // B*K      = 8192
  float* w_curcat = w_cntd + (size_t)B_ * K_;         // B*256    = 65536
  float* w_state  = w_curcat + (size_t)B_ * 256;      // B*H      = 32768
  _Float16* w16   = (_Float16*)(w_state + (size_t)B_ * H_);  // 131072 halves
  _Float16* hW16  = w16 + W16_HID;
  _Float16* pW16  = w16 + W16_PASS;
  _Float16* nW16  = w16 + W16_NEXT;
  _Float16* cW16  = w16 + W16_CAND;

  k_cvt_w<<<W16_TOTAL / 256, 256, 0, stream>>>(hidden_W, pass_W, nexthop_W,
                                               candidate_W, w16);
  k_init<<<dim3(B_ / 16, H_ / 16), 32, 0, stream>>>(start_entities, entity_emb,
                                                    hW16, hidden_b, w_init);
  k_agg_gather<<<B_ * K_, 128, 0, stream>>>(neighbor_nodes, neighbor_relations,
                                            neighbors_num, relation_emb, w_relsum,
                                            w_s0n, w_cntd);
  k_agg_gemm<<<dim3(B_ * K_ / 16, H_ / 16), 32, 0, stream>>>(
      aims, entity_emb, w_init, w_relsum, w_s0n, w_cntd, hW16, pW16, hidden_b,
      pass_b, w_upd);
  k_curcat<<<B_, 256, 0, stream>>>(currents, w_init, w_upd, query, w_curcat);
  k_state_gemm<<<dim3(B_ / 16, H_ / 16), 32, 0, stream>>>(w_curcat, nW16,
                                                          nexthop_b, w_state);
  k_gate<<<1, 256, 0, stream>>>(w_curcat, gate_W, gate_b, out);
  k_cand<<<dim3(MN_ / 128, B_), 256, 0, stream>>>(
      candidate_nodes, candidate_entities, candidate_relations, candidate_masks,
      w_init, w_upd, entity_emb, relation_emb, cW16, candidate_b, w_state, out);
}
